// contrastive_mse_cosinedistance_31353261261533
// MI455X (gfx1250) — matrix operand order repaired and checked by reading, not compiled, not run on hardware
//
#include <hip/hip_runtime.h>

// ---------------------------------------------------------------------------
// Cosine-distance-softmax KL on MI455X (gfx1250, wave32, WMMA bf16 16x16x32).
//
// ~69 GFLOP vs 16 MB HBM traffic -> compute bound. Both 4096x4096 (K=512)
// Gram matrices run on v_wmma_f32_16x16x32_bf16 over bf16-normalized operands
// (8 MB, L2-resident). Softmax shift: cd = 1 - sim, constant cancels;
// -sim in [-1,1] so exp() needs no max subtraction.
//   Z[r]  = sum_j exp(-s[r][j]),  L = log Z
//   KL[r] = sum_j exp(-sx - Lx) * ((sy - sx) + (Ly - Lx))
// K2: LDS-double-buffered B tile filled with GLOBAL_LOAD_ASYNC_TO_LDS_B128
// (ASYNCcnt, no VGPR staging) + register-double-buffered B fragments so DS
// latency hides behind WMMA. K3: register-double-buffered global B streams.
// ---------------------------------------------------------------------------

typedef __attribute__((ext_vector_type(16))) __bf16 v16bf;
typedef __attribute__((ext_vector_type(8)))  float  v8f;

#define NROWS 4096
#define DDIM  512
#define TILE  16
#define KSTEP 32
#define NKF   (DDIM / KSTEP)    // 16 k-fragments per 16x16 output tile
#define NJT   (NROWS / TILE)    // 256 column tiles

union FragU { v16bf v; uint4 q[2]; };

// float -> bf16 round-to-nearest-even
static __device__ inline unsigned short f2bf(float f) {
    union { float f; unsigned int u; } v; v.f = f;
    unsigned int u = v.u;
    return (unsigned short)((u + 0x7FFFu + ((u >> 16) & 1u)) >> 16);
}

// A fragment (16x32 MxK bf16): lane L -> row mbase+(L&15),
// 16 contiguous K at k0+16*(L>>4)
static __device__ inline v16bf load_frag_a(const unsigned short* p, int mbase,
                                           int k0, int lane) {
    int m  = mbase + (lane & 15);
    int kk = k0 + ((lane >> 4) << 4);
    const uint4* q = (const uint4*)(p + (size_t)m * DDIM + kk);
    FragU f;
    f.q[0] = q[0];
    f.q[1] = q[1];
    return f.v;
}

// B fragment (32x16 KxN bf16) from a row-major 16xDDIM tile (B = tile^T):
// lane L -> column (L&15), 16 contiguous K at k0+16*(L>>4)
static __device__ inline v16bf load_frag_b_tile(const unsigned short* tile,
                                                int k0, int lane) {
    int n  = lane & 15;
    int kk = k0 + ((lane >> 4) << 4);
    const uint4* q = (const uint4*)(tile + (size_t)n * DDIM + kk);
    FragU f;
    f.q[0] = q[0];
    f.q[1] = q[1];
    return f.v;
}

static __device__ inline v8f wmma_bf16(v16bf a, v16bf b, v8f c) {
    return __builtin_amdgcn_wmma_f32_16x16x32_bf16(
        false, a, false, b, (short)0, c, false, false);
}

// Async-DMA one 16-row bf16 tile (16 KB) global -> LDS; 128 threads x 8 b128.
// LDS byte address = low 32 bits of the flat pointer (LDS aperture layout).
static __device__ inline void fill_tile_async(unsigned short* dstLds,
                                              const unsigned short* src,
                                              int jbase, int tid) {
    unsigned lds0 = (unsigned)(uintptr_t)dstLds;
    const char* g = (const char*)(src + (size_t)jbase * DDIM);
    #pragma unroll
    for (int i = 0; i < 8; ++i) {
        unsigned           loff = lds0 + (unsigned)(tid + i * 128) * 16u;
        unsigned long long ga   = (unsigned long long)(uintptr_t)(g + (size_t)(tid + i * 128) * 16);
        asm volatile("global_load_async_to_lds_b128 %0, %1, off"
                     :: "v"(loff), "v"(ga) : "memory");
    }
}
static __device__ inline void wait_async0() {
    asm volatile("s_wait_asynccnt 0x0" ::: "memory");
}

// ---------------------------------------------------------------- K1: normalize
__global__ void k_normalize(const float* __restrict__ X, const float* __restrict__ Y,
                            unsigned short* __restrict__ Xb,
                            unsigned short* __restrict__ Yb) {
    int row = blockIdx.x;
    const float* src = blockIdx.y ? Y : X;
    unsigned short* dst = blockIdx.y ? Yb : Xb;
    const float* r = src + (size_t)row * DDIM;
    int tid = threadIdx.x;                      // 256 threads, 2 elems each
    float x0 = r[tid], x1 = r[tid + 256];
    float ss = x0 * x0 + x1 * x1;
    #pragma unroll
    for (int m = 1; m < 32; m <<= 1) ss += __shfl_xor(ss, m, 32);
    __shared__ float red[8];
    if ((tid & 31) == 0) red[tid >> 5] = ss;
    __syncthreads();
    float tot = 0.f;
    #pragma unroll
    for (int i = 0; i < 8; ++i) tot += red[i];
    float inv = 1.0f / fmaxf(sqrtf(tot), 1e-8f);   // eps clamp like reference
    unsigned short* w = dst + (size_t)row * DDIM;
    w[tid]       = f2bf(x0 * inv);
    w[tid + 256] = f2bf(x1 * inv);
}

// ------------------------------------------------- K2: Z[r] = sum_j exp(-s[r][j])
// 4 waves/block, 16 rows per wave. A fragments resident (128 VGPRs). B tiles
// double-buffered in LDS via async DMA; B fragments double-buffered in VGPRs.
__global__ void __launch_bounds__(128, 1)
k_rowsum(const unsigned short* __restrict__ Xb,
         const unsigned short* __restrict__ Yb,
         float* __restrict__ Zx, float* __restrict__ Zy) {
    __shared__ unsigned short bbuf[2][TILE * DDIM];    // 2 x 16 KB
    const unsigned short* P = blockIdx.y ? Yb : Xb;
    float* Z = blockIdx.y ? Zy : Zx;
    int tid = threadIdx.x, wave = tid >> 5, lane = tid & 31;
    int mbase = (blockIdx.x * 4 + wave) * TILE;

    v16bf a[NKF];
    #pragma unroll
    for (int kf = 0; kf < NKF; ++kf) a[kf] = load_frag_a(P, mbase, kf * KSTEP, lane);

    fill_tile_async(bbuf[0], P, 0, tid);
    wait_async0();
    __syncthreads();

    v8f acc = {};
    for (int jt = 0; jt < NJT; ++jt) {
        const unsigned short* cur = bbuf[jt & 1];
        if (jt + 1 < NJT)   // async-DMA next tile while computing this one
            fill_tile_async(bbuf[(jt + 1) & 1], P, (jt + 1) * TILE, tid);
        if (jt + 2 < NJT)
            __builtin_prefetch(P + (size_t)((jt + 2) * TILE + (lane & 15)) * DDIM, 0, 3);
        v8f c = {};
        v16bf b0 = load_frag_b_tile(cur, 0, lane);
        #pragma unroll
        for (int kf = 0; kf < NKF; ++kf) {      // double-buffered B fragments
            v16bf bn = b0;
            if (kf + 1 < NKF) bn = load_frag_b_tile(cur, (kf + 1) * KSTEP, lane);
            c = wmma_bf16(a[kf], b0, c);
            b0 = bn;
        }
        #pragma unroll
        for (int e = 0; e < 8; ++e) acc[e] += __expf(-c[e]);
        wait_async0();                          // next tile landed in LDS
        __syncthreads();
    }
    #pragma unroll
    for (int e = 0; e < 8; ++e) {               // sum the 16 column residues
        float v = acc[e];
        v += __shfl_xor(v, 1, 32); v += __shfl_xor(v, 2, 32);
        v += __shfl_xor(v, 4, 32); v += __shfl_xor(v, 8, 32);
        if ((lane & 15) == 0) Z[mbase + e + ((lane >> 4) << 3)] = v;
    }
}

// ---------------- K3: rowKL[r] = sum_j exp(-sx - Lx) * ((sy - sx) + (Ly - Lx))
// 2 waves/block (32 rows). X A-frags resident; Y rows in LDS (anti-LICM keeps
// them from hoist+spill); global bx/by streams register-double-buffered.
__global__ void __launch_bounds__(64, 1)
k_kl(const unsigned short* __restrict__ Xb,
     const unsigned short* __restrict__ Yb,
     const float* __restrict__ Zx, const float* __restrict__ Zy,
     float* __restrict__ rowkl) {
    __shared__ unsigned short ylds[32 * DDIM];          // 32 KB
    int tid = threadIdx.x;                              // 64 threads
    int wave = tid >> 5, lane = tid & 31;
    int blockRow = blockIdx.x * 32;

    { // async-DMA the block's 32 Y rows into LDS (32 KB = 64 thr x 32 b128)
        unsigned lds0 = (unsigned)(uintptr_t)ylds;
        const char* g = (const char*)(Yb + (size_t)blockRow * DDIM);
        #pragma unroll
        for (int i = 0; i < 32; ++i) {
            unsigned           loff = lds0 + (unsigned)(tid + i * 64) * 16u;
            unsigned long long ga   = (unsigned long long)(uintptr_t)(g + (size_t)(tid + i * 64) * 16);
            asm volatile("global_load_async_to_lds_b128 %0, %1, off"
                         :: "v"(loff), "v"(ga) : "memory");
        }
        wait_async0();
    }
    __syncthreads();

    int mbase = blockRow + wave * TILE;
    int mloc  = wave * TILE;

    v16bf ax[NKF];
    #pragma unroll
    for (int kf = 0; kf < NKF; ++kf) ax[kf] = load_frag_a(Xb, mbase, kf * KSTEP, lane);

    float lx[8], dl[8];
    #pragma unroll
    for (int e = 0; e < 8; ++e) {
        int r = mbase + e + ((lane >> 4) << 3);
        lx[e] = __logf(Zx[r]);
        dl[e] = __logf(Zy[r]) - lx[e];
    }

    int yoff = 0;                 // anti-LICM handle for the LDS ay fragments
    v8f acc = {};
    for (int jt = 0; jt < NJT; ++jt) {
        int jbase = jt * TILE;
        const unsigned short* xrow = Xb + (size_t)jbase * DDIM;
        const unsigned short* yrow = Yb + (size_t)jbase * DDIM;
        if (jt + 1 < NJT) {
            __builtin_prefetch(xrow + (size_t)(TILE + (lane & 15)) * DDIM, 0, 3);
            __builtin_prefetch(yrow + (size_t)(TILE + (lane & 15)) * DDIM, 0, 3);
        }
        asm volatile("" : "+v"(yoff));   // defeat hoisting of the 16 ay frags
        v8f cx = {}, cy = {};
        v16bf bx0 = load_frag_b_tile(xrow, 0, lane);
        v16bf by0 = load_frag_b_tile(yrow, 0, lane);
        #pragma unroll
        for (int kf = 0; kf < NKF; ++kf) {   // double-buffer the global streams
            v16bf bxn = bx0, byn = by0;
            if (kf + 1 < NKF) {
                bxn = load_frag_b_tile(xrow, (kf + 1) * KSTEP, lane);
                byn = load_frag_b_tile(yrow, (kf + 1) * KSTEP, lane);
            }
            cx = wmma_bf16(ax[kf], bx0, cx);
            v16bf ay = load_frag_a(ylds + yoff, mloc, kf * KSTEP, lane); // ds_load
            cy = wmma_bf16(ay, by0, cy);
            bx0 = bxn; by0 = byn;
        }
        #pragma unroll
        for (int e = 0; e < 8; ++e) {
            float sx = cx[e], sy = cy[e];
            float t = __expf(-sx - lx[e]);       // softmax(-s_x) probability
            acc[e] += t * ((sy - sx) + dl[e]);
        }
    }
    #pragma unroll
    for (int e = 0; e < 8; ++e) {
        float v = acc[e];
        v += __shfl_xor(v, 1, 32); v += __shfl_xor(v, 2, 32);
        v += __shfl_xor(v, 4, 32); v += __shfl_xor(v, 8, 32);
        if ((lane & 15) == 0) rowkl[mbase + e + ((lane >> 4) << 3)] = v;
    }
}

// --------------------------------------------------------------- K4: final mean
__global__ void k_reduce(const float* __restrict__ rowkl, float* __restrict__ out) {
    __shared__ float red[8];
    int tid = threadIdx.x;                      // 256 threads
    float s = 0.f;
    for (int i = tid; i < NROWS; i += 256) s += rowkl[i];
    #pragma unroll
    for (int m = 1; m < 32; m <<= 1) s += __shfl_xor(s, m, 32);
    if ((tid & 31) == 0) red[tid >> 5] = s;
    __syncthreads();
    if (tid == 0) {
        float t = 0.f;
        #pragma unroll
        for (int i = 0; i < 8; ++i) t += red[i];
        out[0] = t / (float)NROWS;
    }
}

// ---------------------------------------------------------------------------
extern "C" void kernel_launch(void* const* d_in, const int* in_sizes, int n_in,
                              void* d_out, int out_size, void* d_ws, size_t ws_size,
                              hipStream_t stream) {
    const float* X = (const float*)d_in[0];   // cosine_distance_latent [4096,512] f32
    const float* Y = (const float*)d_in[1];   // mse_latent             [4096,512] f32
    float* out = (float*)d_out;               // scalar f32

    char* ws = (char*)d_ws;                   // layout: Xb(4MB) Yb(4MB) Zx Zy rowkl
    unsigned short* Xb = (unsigned short*)ws;
    unsigned short* Yb = (unsigned short*)(ws + (size_t)NROWS * DDIM * 2);
    float* Zx    = (float*)(ws + (size_t)NROWS * DDIM * 4);
    float* Zy    = Zx + NROWS;
    float* rowkl = Zy + NROWS;

    dim3 g1(NROWS, 2);
    k_normalize<<<g1, 256, 0, stream>>>(X, Y, Xb, Yb);

    dim3 g2(NROWS / (4 * TILE), 2);           // 64 x 2 blocks, 4 waves each
    k_rowsum<<<g2, 128, 0, stream>>>(Xb, Yb, Zx, Zy);

    k_kl<<<NROWS / 32, 64, 0, stream>>>(Xb, Yb, Zx, Zy, rowkl);

    k_reduce<<<1, 256, 0, stream>>>(rowkl, out);
}
